// Gaussian_Attention_3341484557035
// MI455X (gfx1250) — compile-verified, matrix operand order checked by reading, not run on hardware
//
#include <hip/hip_runtime.h>
#include <hip/hip_bf16.h>
#include <stdint.h>

// Problem constants (match reference)
#define B_  2
#define L_  2048
#define E_  512
#define KH  4
#define KE  256
#define NQ  (KH*KE)   // 1024 = K*KE projection width

typedef __attribute__((ext_vector_type(16))) __bf16 v16bf;
typedef __attribute__((ext_vector_type(8)))  float  v8f;

union FragAB {
  v16bf v;
  uint4 q[2];
  unsigned short h[16];
};

static __device__ __forceinline__ unsigned short f2bf(float f) {
  union { float f; unsigned u; } c; c.f = f;
  unsigned u = c.u;
  unsigned r = (u + 0x7FFFu + ((u >> 16) & 1u)) >> 16;   // round-to-nearest-even
  return (unsigned short)r;
}
static __device__ __forceinline__ float bf2f(unsigned short h) {
  union { unsigned u; float f; } c; c.u = ((unsigned)h) << 16;
  return c.f;
}

// ---------------------------------------------------------------------------
// 1) f32 -> bf16 conversions + Wv transpose into (k, f, e) layout
// ---------------------------------------------------------------------------
__global__ void convert_kernel(const float* __restrict__ xsa,
                               const float* __restrict__ Wq,
                               const float* __restrict__ Wk,
                               const float* __restrict__ Wv,
                               unsigned short* __restrict__ xsa_bf,
                               unsigned short* __restrict__ Wq_bf,
                               unsigned short* __restrict__ Wk_bf,
                               unsigned short* __restrict__ WvT_bf) {
  const int n_xsa = B_*L_*E_;      // 2,097,152
  const int n_w   = NQ*E_;         // 524,288
  const int n_wv  = KH*E_*E_;      // 1,048,576
  const int total = n_xsa + 2*n_w + n_wv;
  for (int i = blockIdx.x*blockDim.x + threadIdx.x; i < total;
       i += gridDim.x*blockDim.x) {
    if (i < n_xsa) {
      xsa_bf[i] = f2bf(xsa[i]);
    } else if (i < n_xsa + n_w) {
      int j = i - n_xsa;           Wq_bf[j] = f2bf(Wq[j]);
    } else if (i < n_xsa + 2*n_w) {
      int j = i - n_xsa - n_w;     Wk_bf[j] = f2bf(Wk[j]);
    } else {
      int j = i - n_xsa - 2*n_w;   // output index into WvT[k][f][e]
      int k = j >> 18;             // / (512*512)
      int r = j & (E_*E_ - 1);
      int f = r >> 9, e = r & 511;
      WvT_bf[j] = f2bf(Wv[(k << 18) + e*E_ + f]);   // Wv3[k][e][f]
    }
  }
}

// ---------------------------------------------------------------------------
// 2) Projection GEMM: Out(4096x1024 bf16) = A(4096x512 bf16) @ Wb(1024x512)^T
//    4 waves, each owning a 64x64 tile (4x4 WMMA grid) -> 16 WMMAs per
//    8 fragment loads per k-step (~64 flop/byte from L2).
//    Optional per-feature bias biasW[d*KH + k] (Gaussian center fold for K).
// ---------------------------------------------------------------------------
__global__ void __launch_bounds__(128)
proj_gemm(const unsigned short* __restrict__ A,
          const unsigned short* __restrict__ Wb,
          const float* __restrict__ biasW,
          unsigned short* __restrict__ Out) {
  const int lane = threadIdx.x & 31;
  const int wv   = threadIdx.x >> 5;            // 4 waves, 2x2 tiling
  const int wm = wv >> 1, wn = wv & 1;
  const int mb = blockIdx.x*128 + wm*64;
  const int nb = blockIdx.y*128 + wn*64;
  const int hi = lane >> 4, lm = lane & 15;

  v8f acc[4][4];
  for (int i = 0; i < 4; ++i) for (int j = 0; j < 4; ++j) { v8f z = {}; acc[i][j] = z; }

  for (int kb = 0; kb < E_; kb += 32) {
    FragAB a[4], b[4];
    for (int t = 0; t < 4; ++t) {
      const unsigned short* ap = A  + (size_t)(mb + 16*t + lm)*E_ + kb + 8*hi;
      a[t].q[0] = *(const uint4*)ap;
      a[t].q[1] = *(const uint4*)(ap + 16);
      const unsigned short* bp = Wb + (size_t)(nb + 16*t + lm)*E_ + kb + 8*hi;
      b[t].q[0] = *(const uint4*)bp;
      b[t].q[1] = *(const uint4*)(bp + 16);
      if (kb + 32 < E_) {                     // gfx1250 global_prefetch_b8
        __builtin_prefetch(ap + 32, 0, 1);
        __builtin_prefetch(bp + 32, 0, 1);
      }
    }
    for (int i = 0; i < 4; ++i)
      for (int j = 0; j < 4; ++j)
        acc[i][j] = __builtin_amdgcn_wmma_f32_16x16x32_bf16(
            false, a[i].v, false, b[j].v, (short)0, acc[i][j], false, false);
  }

  for (int i = 0; i < 4; ++i) {
    for (int j = 0; j < 4; ++j) {
      const int feat = nb + 16*j + lm;
      float bias = 0.0f;
      if (biasW) bias = biasW[(feat & (KE-1))*KH + (feat >> 8)];  // biasW[d][k]
      for (int v = 0; v < 8; ++v) {
        const int row = mb + 16*i + v + 8*hi;
        Out[(size_t)row*NQ + feat] = f2bf(acc[i][j][v] + bias);
      }
    }
  }
}

// ---------------------------------------------------------------------------
// 3) Row norms: Out[(b*L+l)*KH + k] = sum_d X[bl][k*KE+d]^2   (from bf16)
// ---------------------------------------------------------------------------
__global__ void norms_kernel(const unsigned short* __restrict__ X,
                             float* __restrict__ Out) {
  const int idx = blockIdx.x*blockDim.x + threadIdx.x;
  if (idx >= B_*L_*KH) return;
  const unsigned short* p = X + (size_t)(idx >> 2)*NQ + (idx & 3)*KE;
  float s = 0.0f;
  for (int d = 0; d < KE; d += 8) {
    uint4 q = *(const uint4*)(p + d);
    const unsigned short* hh = (const unsigned short*)&q;
    for (int t = 0; t < 8; ++t) { float f = bf2f(hh[t]); s += f*f; }
  }
  Out[idx] = s;
}

// ---------------------------------------------------------------------------
// 4) Flash-style Gaussian attention per (b, head, 16-query tile).
//    h_ij = (2*q.kb - qn_i - kn_j) / sqrt(KE); online softmax; yik = P @ xsa.
// ---------------------------------------------------------------------------
#define BKV 64
#define SS  72   // padded LDS row stride (stride 36 banks mod 64 -> conflict-free)

__global__ void __launch_bounds__(128)
flash_kernel(const unsigned short* __restrict__ xq,
             const unsigned short* __restrict__ xkb,
             const unsigned short* __restrict__ xsa_bf,
             const float* __restrict__ qn,
             const float* __restrict__ kn,
             unsigned short* __restrict__ yik) {
  __shared__ unsigned short VT[256*SS];   // half of E, transposed: VT[e_local][j]
  __shared__ float          Sb[16*SS];    // score tile (raw q.kb dot products)
  __shared__ unsigned short Pb[16*SS];    // probability tile, bf16
  __shared__ float m_run[16], l_run[16], alpha_s[16], qn_s[16];
  __shared__ float red[16*8];

  const int qt = blockIdx.x, kh = blockIdx.y, b = blockIdx.z;
  const int qbase = qt*16;
  const int t = threadIdx.x;
  const int lane = t & 31, wv = t >> 5;
  const int hi = lane >> 4, lm = lane & 15;

  // Preload A fragments (xq tile 16x256) once -- invariant over the key loop.
  FragAB afrag[8];
  {
    const unsigned short* ap = xq + (size_t)(b*L_ + qbase + lm)*NQ + kh*KE;
    for (int c = 0; c < 8; ++c) {
      const unsigned short* p = ap + 32*c + 8*hi;
      afrag[c].q[0] = *(const uint4*)p;
      afrag[c].q[1] = *(const uint4*)(p + 16);
    }
  }
  v8f acc[8];
  for (int i = 0; i < 8; ++i) { v8f z = {}; acc[i] = z; }

  if (t < 16) {
    m_run[t] = -3.0e38f;
    l_run[t] = 0.0f;
    qn_s[t]  = qn[(size_t)(b*L_ + qbase + t)*KH + kh];
  }
  __syncthreads();

  const float rs = 0.0625f;  // 1/sqrt(KE)

  for (int jb = 0; jb < L_; jb += BKV) {
    // ---- scores: wave wv computes keys [jb+16*wv, jb+16*wv+16)
    {
      v8f sfrag = {};
      const unsigned short* bp0 =
          xkb + (size_t)(b*L_ + jb + 16*wv + lm)*NQ + kh*KE + 8*hi;
      for (int c = 0; c < 8; ++c) {
        FragAB bf;
        bf.q[0] = *(const uint4*)(bp0 + 32*c);
        bf.q[1] = *(const uint4*)(bp0 + 32*c + 16);
        sfrag = __builtin_amdgcn_wmma_f32_16x16x32_bf16(
            false, afrag[c].v, false, bf.v, (short)0, sfrag, false, false);
      }
      for (int v = 0; v < 8; ++v)
        Sb[(v + 8*hi)*SS + 16*wv + lm] = sfrag[v];
    }
    __syncthreads();

    // ---- online softmax over this 16x64 tile
    {
      const int r = t >> 3, c0 = (t & 7)*8;
      const float qr = qn_s[r];
      float hloc[8];
      float lmax = -3.0e38f;
      for (int c = 0; c < 8; ++c) {
        const int j = jb + c0 + c;
        const float s = Sb[r*SS + c0 + c];
        const float h = (2.0f*s - qr - kn[(size_t)(b*L_ + j)*KH + kh]) * rs;
        hloc[c] = h;
        lmax = fmaxf(lmax, h);
      }
      red[r*8 + (t & 7)] = lmax;
      __syncthreads();
      if ((t & 7) == 0) {
        float m = red[r*8];
        for (int u = 1; u < 8; ++u) m = fmaxf(m, red[r*8 + u]);
        const float mold = m_run[r];
        const float mnew = fmaxf(mold, m);
        alpha_s[r] = __expf(mold - mnew);
        m_run[r]   = mnew;
      }
      __syncthreads();
      const float mnew = m_run[r];
      float lsum = 0.0f;
      for (int c = 0; c < 8; ++c) {
        const float p = __expf(hloc[c] - mnew);
        lsum += p;
        Pb[r*SS + c0 + c] = f2bf(p);
      }
      red[r*8 + (t & 7)] = lsum;
      __syncthreads();
      if ((t & 7) == 0) {
        float s8 = 0.0f;
        for (int u = 0; u < 8; ++u) s8 += red[r*8 + u];
        l_run[r] = alpha_s[r]*l_run[r] + s8;
      }
    }
    __syncthreads();

    // ---- rescale accumulators by alpha (per query row)
    for (int v = 0; v < 8; ++v) {
      const float al = alpha_s[v + 8*hi];
      for (int i = 0; i < 8; ++i) acc[i][v] *= al;
    }

    // ---- P @ V in two E halves (stage V transposed in LDS)
    for (int half = 0; half < 2; ++half) {
      // vectorized staging: b128 global reads (8 bf16, contiguous e),
      // b16 LDS scatter stores performing the transpose
      for (int i8 = t; i8 < BKV*32; i8 += 128) {
        const int j  = i8 >> 5;          // key 0..63
        const int ec = (i8 & 31) * 8;    // e_local chunk base
        uint4 qv = *(const uint4*)
            &xsa_bf[(size_t)(b*L_ + jb + j)*E_ + half*256 + ec];
        const unsigned short* hh = (const unsigned short*)&qv;
#pragma unroll
        for (int u = 0; u < 8; ++u) VT[(ec + u)*SS + j] = hh[u];
      }
      __syncthreads();
      for (int kc = 0; kc < BKV; kc += 32) {
        FragAB pf;
        {
          const unsigned short* pp = &Pb[lm*SS + kc + 8*hi];
          pf.q[0] = *(const uint4*)pp;
          pf.q[1] = *(const uint4*)(pp + 16);
        }
        for (int et = 0; et < 4; ++et) {
          const int el = wv*64 + et*16 + lm;     // e_local for this lane
          FragAB bf;
          const unsigned short* vp = &VT[el*SS + kc + 8*hi];
          bf.q[0] = *(const uint4*)vp;
          bf.q[1] = *(const uint4*)(vp + 16);
          acc[half*4 + et] = __builtin_amdgcn_wmma_f32_16x16x32_bf16(
              false, pf.v, false, bf.v, (short)0, acc[half*4 + et], false, false);
        }
      }
      __syncthreads();
    }
  }

  // ---- epilogue: normalize by l, store yik as bf16
  for (int v = 0; v < 8; ++v) {
    const int r = v + 8*hi;
    const float rl = 1.0f / l_run[r];
    const int row = b*L_ + qbase + r;
    for (int i = 0; i < 8; ++i) {
      const int e = (i >> 2)*256 + wv*64 + (i & 3)*16 + lm;
      yik[((size_t)row*KH + kh)*E_ + e] = f2bf(acc[i][v] * rl);
    }
  }
}

// ---------------------------------------------------------------------------
// 5) Output GEMM per head: out[m, kh*E+f] = sum_e yik[m,kh,e]*WvT[kh,f,e] + b2
//    Same 64x64-per-wave register blocking as proj_gemm.
// ---------------------------------------------------------------------------
__global__ void __launch_bounds__(128)
out_gemm(const unsigned short* __restrict__ yik,
         const unsigned short* __restrict__ WvT,
         const float* __restrict__ bias2,
         float* __restrict__ out) {
  const int lane = threadIdx.x & 31;
  const int wvd  = threadIdx.x >> 5;
  const int wm = wvd >> 1, wn = wvd & 1;
  const int kh = blockIdx.z;
  const int mb = blockIdx.x*128 + wm*64;
  const int nb = blockIdx.y*128 + wn*64;
  const int hi = lane >> 4, lm = lane & 15;

  const unsigned short* A  = yik + (size_t)kh*E_;        // + m*(KH*E_)
  const unsigned short* Bw = WvT + (size_t)kh*E_*E_;     // (f, e) row-major

  v8f acc[4][4];
  for (int i = 0; i < 4; ++i) for (int j = 0; j < 4; ++j) { v8f z = {}; acc[i][j] = z; }

  for (int kb = 0; kb < E_; kb += 32) {
    FragAB a[4], bfr[4];
    for (int tt = 0; tt < 4; ++tt) {
      const unsigned short* ap = A  + (size_t)(mb + 16*tt + lm)*(KH*E_) + kb + 8*hi;
      a[tt].q[0] = *(const uint4*)ap;
      a[tt].q[1] = *(const uint4*)(ap + 16);
      const unsigned short* bp = Bw + (size_t)(nb + 16*tt + lm)*E_ + kb + 8*hi;
      bfr[tt].q[0] = *(const uint4*)bp;
      bfr[tt].q[1] = *(const uint4*)(bp + 16);
      if (kb + 32 < E_) {
        __builtin_prefetch(ap + 32, 0, 1);
        __builtin_prefetch(bp + 32, 0, 1);
      }
    }
    for (int i = 0; i < 4; ++i)
      for (int j = 0; j < 4; ++j)
        acc[i][j] = __builtin_amdgcn_wmma_f32_16x16x32_bf16(
            false, a[i].v, false, bfr[j].v, (short)0, acc[i][j], false, false);
  }

  for (int j = 0; j < 4; ++j) {
    const int f = nb + 16*j + lm;
    const float bv = bias2[f*KH + kh];   // bias2W[f][k]
    for (int i = 0; i < 4; ++i) {
      for (int v = 0; v < 8; ++v) {
        const int row = mb + 16*i + v + 8*hi;
        out[(size_t)row*(KH*E_) + kh*E_ + f] = acc[i][j][v] + bv;
      }
    }
  }
}

// ---------------------------------------------------------------------------
extern "C" void kernel_launch(void* const* d_in, const int* in_sizes, int n_in,
                              void* d_out, int out_size, void* d_ws, size_t ws_size,
                              hipStream_t stream) {
  (void)in_sizes; (void)n_in; (void)out_size; (void)ws_size;
  const float* xsa   = (const float*)d_in[0];
  const float* Wq    = (const float*)d_in[1];
  const float* Wk    = (const float*)d_in[2];
  const float* Wv    = (const float*)d_in[3];
  const float* biasW = (const float*)d_in[4];
  const float* bias2 = (const float*)d_in[5];
  float* out = (float*)d_out;

  char* ws = (char*)d_ws;
  size_t off = 0;
  auto alloc = [&](size_t bytes) -> char* {
    char* p = ws + off;
    off += (bytes + 255) & ~(size_t)255;
    return p;
  };
  unsigned short* xsa_bf = (unsigned short*)alloc((size_t)B_*L_*E_*2);   // 4 MB
  unsigned short* Wq_bf  = (unsigned short*)alloc((size_t)NQ*E_*2);      // 1 MB
  unsigned short* Wk_bf  = (unsigned short*)alloc((size_t)NQ*E_*2);      // 1 MB
  unsigned short* WvT_bf = (unsigned short*)alloc((size_t)KH*E_*E_*2);   // 2 MB
  unsigned short* xq_bf  = (unsigned short*)alloc((size_t)B_*L_*NQ*2);   // 8 MB
  unsigned short* xkb_bf = (unsigned short*)alloc((size_t)B_*L_*NQ*2);   // 8 MB
  float*          qn     = (float*)alloc((size_t)B_*L_*KH*4);
  float*          kn     = (float*)alloc((size_t)B_*L_*KH*4);
  unsigned short* yik_bf = (unsigned short*)alloc((size_t)B_*L_*KH*E_*2); // 16 MB

  convert_kernel<<<4096, 256, 0, stream>>>(xsa, Wq, Wk, Wv,
                                           xsa_bf, Wq_bf, Wk_bf, WvT_bf);

  // Q / K projections: M=4096, N=1024, K=512, block tile 128x128
  proj_gemm<<<dim3(32, 8), 128, 0, stream>>>(xsa_bf, Wq_bf, nullptr, xq_bf);
  proj_gemm<<<dim3(32, 8), 128, 0, stream>>>(xsa_bf, Wk_bf, biasW,   xkb_bf);

  norms_kernel<<<64, 256, 0, stream>>>(xq_bf,  qn);
  norms_kernel<<<64, 256, 0, stream>>>(xkb_bf, kn);

  // Flash Gaussian attention: grid = (L/16 query tiles, heads, batch)
  flash_kernel<<<dim3(L_/16, KH, B_), 128, 0, stream>>>(
      xq_bf, xkb_bf, xsa_bf, qn, kn, yik_bf);

  // Per-head output GEMM: M=4096, N=512 per head, block tile 128x128
  out_gemm<<<dim3(32, 4, KH), 128, 0, stream>>>(yik_bf, WvT_bf, bias2, out);
}
